// NeuralCellularAutomata2_78211354460279
// MI455X (gfx1250) — matrix operand order repaired and checked by reading, not compiled, not run on hardware
//
#include <hip/hip_runtime.h>
#include <hip/hip_bf16.h>
#include <math.h>

// B=16, H=64, W=64, C=256 ; N = 65536 pixels
#define NPIX   65536
#define CCH    256

typedef __attribute__((ext_vector_type(16))) __bf16 v16bf;
typedef __attribute__((ext_vector_type(8)))  float  v8f;

union FragU { uint4 u4[2]; v16bf v; };

// gfx1250 async HBM->LDS path (ASYNCcnt-tracked), with safe fallback.
#if defined(__AMDGCN__) && __has_builtin(__builtin_amdgcn_global_load_async_to_lds_b128)
#define ASYNC_LDS 1
typedef __attribute__((__vector_size__(16))) int v4i_g;
typedef __attribute__((address_space(1))) v4i_g* as1_v4i;
typedef __attribute__((address_space(3))) v4i_g* as3_v4i;

__device__ __forceinline__ void async_ld_b128(const void* g, void* l) {
  // AS1 ptr is 64-bit; AS3 ptr is the low-32 LDS byte address of the generic
  // shared pointer (upper aperture bits discarded per ISA).
  as1_v4i gp = (as1_v4i)(unsigned long long)g;
  as3_v4i lp = (as3_v4i)(unsigned int)(unsigned long long)l;
  __builtin_amdgcn_global_load_async_to_lds_b128(gp, lp, 0, 0);
}
#else
#define ASYNC_LDS 0
#endif

__device__ __forceinline__ void wait_async_lds() {
#if ASYNC_LDS
#if __has_builtin(__builtin_amdgcn_s_wait_asynccnt)
  __builtin_amdgcn_s_wait_asynccnt(0);
#else
  asm volatile("s_wait_asynccnt 0" ::: "memory");
#endif
#endif
}

__device__ __forceinline__ unsigned short f2bf(float f) {
  unsigned int u = __float_as_uint(f);
  unsigned int r = u + 0x7fffu + ((u >> 16) & 1u);   // round-to-nearest-even
  return (unsigned short)(r >> 16);
}

// ---------------------------------------------------------------------------
// fp32 -> bf16 elementwise convert (for weights)
// ---------------------------------------------------------------------------
__global__ void cvt_bf16_kernel(const float* __restrict__ src,
                                unsigned short* __restrict__ dst, int n) {
  int i = blockIdx.x * blockDim.x + threadIdx.x;
  if (i < n) dst[i] = f2bf(src[i]);
}

// ---------------------------------------------------------------------------
// Depthwise 3x3 conv (groups=C, 3 outputs per group), NHWC input.
// perceived[n][d] , d = 3*c + r , input channel = d/3. Output bf16.
// ---------------------------------------------------------------------------
__global__ void __launch_bounds__(768)
dwconv_kernel(const float* __restrict__ h, const float* __restrict__ wp,
              const float* __restrict__ bp, unsigned short* __restrict__ perc) {
  const int n = blockIdx.x;
  const int d = threadIdx.x;            // 0..767
  const int c = d / 3;                  // group = input channel
  const int b = n >> 12;
  const int y = (n >> 6) & 63;
  const int x = n & 63;
  float acc = bp[d];
  const float* wd = wp + d * 9;
  #pragma unroll
  for (int ky = 0; ky < 3; ++ky) {
    const int yy = y + ky - 1;
    #pragma unroll
    for (int kx = 0; kx < 3; ++kx) {
      const int xx = x + kx - 1;
      float xv = 0.0f;
      if ((unsigned)yy < 64u && (unsigned)xx < 64u) {
        const int nb = ((b * 64 + yy) * 64 + xx);
        xv = h[(size_t)nb * CCH + c];
      }
      acc += xv * wd[ky * 3 + kx];
    }
  }
  perc[(size_t)n * 768 + d] = f2bf(acc);
}

// ---------------------------------------------------------------------------
// bf16 WMMA GEMM:  C[M,N] = A[M,K] * Bt[N,K]^T + bias[N]   (all tiles exact)
// Block = 256 thr (8 waves), tile 128x128, K-step 32.
// MODE 0: GELU(exact) -> bf16 out             (hid)
// MODE 1: + resid fp32 -> fp32 out + bf16 out (h_new)
// MODE 2: -> fp32 out                         (qkv)
// ---------------------------------------------------------------------------
template <int MODE>
__global__ void __launch_bounds__(256)
gemm_wmma_kernel(const unsigned short* __restrict__ A,
                 const unsigned short* __restrict__ Bt,
                 const float* __restrict__ bias,
                 int M, int K, int N,
                 const float* __restrict__ resid,
                 float* __restrict__ outF,
                 unsigned short* __restrict__ outB) {
  __shared__ unsigned short sA[128 * 40];   // 32 k-cols + 8 pad
  __shared__ unsigned short sB[128 * 40];

  const int tid  = threadIdx.x;
  const int lane = tid & 31;
  const int wid  = tid >> 5;         // 0..7
  const int wm   = wid >> 1;         // M wave offset: 32*wm
  const int wn   = wid & 1;          // N wave offset: 64*wn
  const int mloc = lane & 15;
  const int half = lane >> 4;

  const int nBlkN = N >> 7;
  const int row0  = (blockIdx.x / nBlkN) << 7;
  const int col0  = (blockIdx.x % nBlkN) << 7;

  const int lr = tid & 127;          // staging row
  const int ls = tid >> 7;           // staging 16-elem segment (0/1)

  v8f zero = {0.f, 0.f, 0.f, 0.f, 0.f, 0.f, 0.f, 0.f};
  v8f acc[2][4];
  #pragma unroll
  for (int im = 0; im < 2; ++im)
    #pragma unroll
    for (int in = 0; in < 4; ++in) acc[im][in] = zero;

  for (int kt = 0; kt < K; kt += 32) {
    { // stage A [128 x 32] and Bt [128 x 32] tiles into LDS
      const unsigned short* ga = A  + (size_t)(row0 + lr) * K + kt + 16 * ls;
      const unsigned short* gb = Bt + (size_t)(col0 + lr) * K + kt + 16 * ls;
      unsigned short* la = &sA[lr * 40 + 16 * ls];
      unsigned short* lb = &sB[lr * 40 + 16 * ls];
#if ASYNC_LDS
      // HBM -> LDS direct (global_load_async_to_lds_b128, ASYNCcnt-tracked)
      async_ld_b128(ga,     la);
      async_ld_b128(ga + 8, la + 8);
      async_ld_b128(gb,     lb);
      async_ld_b128(gb + 8, lb + 8);
#else
      uint4 a0 = *(const uint4*)(ga);
      uint4 a1 = *(const uint4*)(ga + 8);
      uint4 b0 = *(const uint4*)(gb);
      uint4 b1 = *(const uint4*)(gb + 8);
      *(uint4*)(la)     = a0;
      *(uint4*)(la + 8) = a1;
      *(uint4*)(lb)     = b0;
      *(uint4*)(lb + 8) = b1;
#endif
    }
    wait_async_lds();
    __syncthreads();

    if (kt + 32 < K) {  // prefetch next K tile -> global_prefetch_b8
      __builtin_prefetch(A  + (size_t)(row0 + lr) * K + kt + 32 + 16 * ls, 0, 1);
      __builtin_prefetch(Bt + (size_t)(col0 + lr) * K + kt + 32 + 16 * ls, 0, 1);
    }

    // A fragment (16x32 bf16): lane m=lane%16; elems 0..7: K=8*half+0..7,
    // elems 8..15: K=16+8*half+0..7  (ISA 7.12.2)
    FragU fa[2];
    #pragma unroll
    for (int im = 0; im < 2; ++im) {
      const int r = 32 * wm + 16 * im + mloc;
      fa[im].u4[0] = *(const uint4*)(&sA[r * 40 + 8 * half]);
      fa[im].u4[1] = *(const uint4*)(&sA[r * 40 + 16 + 8 * half]);
    }
    // B fragment (32x16 bf16): lane n=lane%16; elems 0..7: K=8*half+0..7, elems 8..15: K=16+8*half+0..7
    FragU fb[4];
    #pragma unroll
    for (int in = 0; in < 4; ++in) {
      const int r = 64 * wn + 16 * in + mloc;
      fb[in].u4[0] = *(const uint4*)(&sB[r * 40 + 8 * half]);
      fb[in].u4[1] = *(const uint4*)(&sB[r * 40 + 16 + 8 * half]);
    }

    #pragma unroll
    for (int im = 0; im < 2; ++im)
      #pragma unroll
      for (int in = 0; in < 4; ++in)
        acc[im][in] = __builtin_amdgcn_wmma_f32_16x16x32_bf16(
            false, fa[im].v, false, fb[in].v, (short)0, acc[im][in],
            false, false);
    __syncthreads();
  }

  // Epilogue. C/D layout: col = lane%16, row(v) = v + 8*half (ISA 7.12.2)
  #pragma unroll
  for (int im = 0; im < 2; ++im) {
    #pragma unroll
    for (int in = 0; in < 4; ++in) {
      const int ncol = col0 + 64 * wn + 16 * in + mloc;
      const float bv = bias[ncol];
      #pragma unroll
      for (int v = 0; v < 8; ++v) {
        const int mrow = row0 + 32 * wm + 16 * im + v + 8 * half;
        const size_t idx = (size_t)mrow * N + ncol;
        float val = acc[im][in][v] + bv;
        if (MODE == 0) {
          float g = 0.5f * val * (1.0f + erff(val * 0.70710678118654752f));
          outB[idx] = f2bf(g);
        } else if (MODE == 1) {
          float hn = val + resid[idx];
          outF[idx] = hn;
          outB[idx] = f2bf(hn);
        } else {
          outF[idx] = val;
        }
      }
    }
  }
}

// ---------------------------------------------------------------------------
// 3x3 windowed local attention; one wave32 per pixel, lane owns 8 channels.
// qkv fp32 [N,768] (q|k|v). Border taps are zero-padded k/v => score 0,
// v-contribution 0 (matches reference). out += attention.
// ---------------------------------------------------------------------------
__global__ void __launch_bounds__(256)
attn_kernel(const float* __restrict__ qkv, float* __restrict__ out) {
  const int wid  = threadIdx.x >> 5;
  const int lane = threadIdx.x & 31;
  const int n = blockIdx.x * 8 + wid;
  const int b = n >> 12, y = (n >> 6) & 63, x = n & 63;
  const int c0 = lane * 8;

  const float* qp = qkv + (size_t)n * 768 + c0;
  const float4 q0 = *(const float4*)(qp);
  const float4 q1 = *(const float4*)(qp + 4);
  const float scale = 0.0625f;  // 1/sqrt(256)

  float s[9];
  int   nn[9];
  #pragma unroll
  for (int i = 0; i < 9; ++i) {
    const int dy = i / 3 - 1, dx = i % 3 - 1;
    const int yy = y + dy, xx = x + dx;
    const bool valid = ((unsigned)yy < 64u) && ((unsigned)xx < 64u);
    nn[i] = valid ? ((b * 64 + yy) * 64 + xx) : -1;
    float p = 0.0f;
    if (valid) {
      const float* kp = qkv + (size_t)nn[i] * 768 + 256 + c0;
      const float4 k0 = *(const float4*)(kp);
      const float4 k1 = *(const float4*)(kp + 4);
      p = q0.x * k0.x + q0.y * k0.y + q0.z * k0.z + q0.w * k0.w +
          q1.x * k1.x + q1.y * k1.y + q1.z * k1.z + q1.w * k1.w;
    }
    #pragma unroll
    for (int m = 16; m >= 1; m >>= 1) p += __shfl_xor(p, m, 32);
    s[i] = p * scale;   // 0 for zero-padded border taps
  }

  float mx = s[0];
  #pragma unroll
  for (int i = 1; i < 9; ++i) mx = fmaxf(mx, s[i]);
  float w[9], sum = 0.0f;
  #pragma unroll
  for (int i = 0; i < 9; ++i) { w[i] = __expf(s[i] - mx); sum += w[i]; }
  const float inv = 1.0f / sum;

  float a0x = 0, a0y = 0, a0z = 0, a0w = 0, a1x = 0, a1y = 0, a1z = 0, a1w = 0;
  #pragma unroll
  for (int i = 0; i < 9; ++i) {
    if (nn[i] >= 0) {
      const float* vp = qkv + (size_t)nn[i] * 768 + 512 + c0;
      const float4 v0 = *(const float4*)(vp);
      const float4 v1 = *(const float4*)(vp + 4);
      const float wi = w[i] * inv;
      a0x += wi * v0.x; a0y += wi * v0.y; a0z += wi * v0.z; a0w += wi * v0.w;
      a1x += wi * v1.x; a1y += wi * v1.y; a1z += wi * v1.z; a1w += wi * v1.w;
    }
  }

  float* op = out + (size_t)n * CCH + c0;
  float4 o0 = *(float4*)(op);
  float4 o1 = *(float4*)(op + 4);
  o0.x += a0x; o0.y += a0y; o0.z += a0z; o0.w += a0w;
  o1.x += a1x; o1.y += a1y; o1.z += a1z; o1.w += a1w;
  *(float4*)(op)     = o0;
  *(float4*)(op + 4) = o1;
}

// ---------------------------------------------------------------------------
extern "C" void kernel_launch(void* const* d_in, const int* in_sizes, int n_in,
                              void* d_out, int out_size, void* d_ws, size_t ws_size,
                              hipStream_t stream) {
  const float* h      = (const float*)d_in[0];
  const float* w_perc = (const float*)d_in[1];
  const float* b_perc = (const float*)d_in[2];
  const float* w_up1  = (const float*)d_in[3];   // [512,768] = [N,K]
  const float* b_up1  = (const float*)d_in[4];
  const float* w_up2  = (const float*)d_in[5];   // [256,512]
  const float* b_up2  = (const float*)d_in[6];
  const float* w_qkv  = (const float*)d_in[7];   // [768,256]
  const float* b_qkv  = (const float*)d_in[8];
  float* out = (float*)d_out;

  char* ws = (char*)d_ws;
  size_t off = 0;
  auto take = [&](size_t bytes) -> char* {
    char* p = ws + off;
    off = (off + bytes + 255) & ~(size_t)255;
    return p;
  };
  unsigned short* wu1b  = (unsigned short*)take((size_t)512 * 768 * 2);
  unsigned short* wu2b  = (unsigned short*)take((size_t)256 * 512 * 2);
  unsigned short* wqkvb = (unsigned short*)take((size_t)768 * 256 * 2);
  unsigned short* percb = (unsigned short*)take((size_t)NPIX * 768 * 2);
  unsigned short* hidb  = (unsigned short*)take((size_t)NPIX * 512 * 2);
  unsigned short* hnewb = (unsigned short*)take((size_t)NPIX * 256 * 2);
  float*          qkvf  = (float*)take((size_t)NPIX * 768 * 4);
  (void)ws_size; (void)in_sizes; (void)n_in; (void)out_size;

  // 1) weights -> bf16 (layout already [N,K])
  cvt_bf16_kernel<<<(512 * 768 + 255) / 256, 256, 0, stream>>>(w_up1, wu1b, 512 * 768);
  cvt_bf16_kernel<<<(256 * 512 + 255) / 256, 256, 0, stream>>>(w_up2, wu2b, 256 * 512);
  cvt_bf16_kernel<<<(768 * 256 + 255) / 256, 256, 0, stream>>>(w_qkv, wqkvb, 768 * 256);

  // 2) depthwise perception conv -> perceived bf16 [N,768]
  dwconv_kernel<<<NPIX, 768, 0, stream>>>(h, w_perc, b_perc, percb);

  // 3) GEMM1: [N,768]x[768,512] +bias, GELU -> hid bf16
  gemm_wmma_kernel<0><<<(NPIX / 128) * (512 / 128), 256, 0, stream>>>(
      percb, wu1b, b_up1, NPIX, 768, 512, nullptr, nullptr, hidb);

  // 4) GEMM2: [N,512]x[512,256] +bias, +h residual -> d_out fp32 & h_new bf16
  gemm_wmma_kernel<1><<<(NPIX / 128) * (256 / 128), 256, 0, stream>>>(
      hidb, wu2b, b_up2, NPIX, 512, 256, h, out, hnewb);

  // 5) GEMM3: [N,256]x[256,768] +bias -> qkv fp32
  gemm_wmma_kernel<2><<<(NPIX / 128) * (768 / 128), 256, 0, stream>>>(
      hnewb, wqkvb, b_qkv, NPIX, 256, 768, nullptr, qkvf, nullptr);

  // 6) local attention, out += h_attn
  attn_kernel<<<NPIX / 8, 256, 0, stream>>>(qkvf, out);
}
